// Model_18786186952982
// MI455X (gfx1250) — matrix operand order repaired and checked by reading, not compiled, not run on hardware
//
#include <hip/hip_runtime.h>
#include <cstddef>
#include <cstdint>

// ---------------------------------------------------------------------------
// MI455X (gfx1250) implementation of the MeshGraphNet-style reference.
// All GEMMs (encoders, 15x(4 edge MLPs + node MLP), decoder layers 1-2) run on
// v_wmma_f32_16x16x32_bf16 with f32 accumulation; weights are pre-converted to
// bf16 in a transposed layout Wt[n][k] so B-fragments are contiguous b128
// loads.  Activations kept as f32 masters + bf16 mirrors for WMMA operands.
// ---------------------------------------------------------------------------

typedef __attribute__((ext_vector_type(16))) __bf16 v16bf;
typedef __attribute__((ext_vector_type(8)))  float  v8f;
typedef __attribute__((ext_vector_type(2)))  unsigned long long ull2;

union BF16x16 { v16bf v; ull2 q[2]; };
union BF16x8  { __bf16 h[8]; ull2 q; };

__device__ __forceinline__ __bf16 f2bf(float x) {
  union { float f; unsigned u; } a; a.f = x;
  unsigned r = a.u + 0x7FFFu + ((a.u >> 16) & 1u);   // round-to-nearest-even
  unsigned short hs = (unsigned short)(r >> 16);
  union { unsigned short s; __bf16 b; } y; y.s = hs; return y.b;
}
__device__ __forceinline__ float bf2f(__bf16 b) {
  union { unsigned short s; __bf16 b; } y; y.b = b;
  union { unsigned u; float f; } a; a.u = ((unsigned)y.s) << 16; return a.f;
}

// 16x16 output tile of X[16 x kp] @ Wt^T (Wt stored [N][kp], bf16), f32 accum.
// lane layout per CDNA5 ISA 7.12.2 (wave32).
__device__ __forceinline__ v8f wmma_tile(const __bf16* Xs, int kp,
                                         const __bf16* Wt, int lane, int gcol) {
  v8f acc = {0.f, 0.f, 0.f, 0.f, 0.f, 0.f, 0.f, 0.f};
  const int row = lane & 15;
  const int akb = (lane & 16) ? 8 : 0;    // A: K sub-block per lane half
  const int bkb = (lane & 16) ? 8 : 0;    // B: K sub-block per lane half
  const __bf16* ap = Xs + (size_t)row * kp + akb;
  const __bf16* bp = Wt + (size_t)gcol * kp + bkb;
  for (int kc = 0; kc < kp; kc += 32) {
    BF16x16 a, b;
    a.q[0] = *(const ull2*)(ap + kc);        // K = akb .. akb+7
    a.q[1] = *(const ull2*)(ap + kc + 16);   // K = akb+16 .. akb+23
    b.q[0] = *(const ull2*)(bp + kc);        // K = bkb .. bkb+7
    b.q[1] = *(const ull2*)(bp + kc + 16);   // K = bkb+16 .. bkb+23
    acc = __builtin_amdgcn_wmma_f32_16x16x32_bf16(false, a.v, false, b.v,
                                                  (short)0, acc, false, false);
  }
  return acc;
}

__device__ __forceinline__ void layer_relu(const __bf16* Xs, int kp,
                                           const __bf16* Wt, const float* B,
                                           __bf16* Xn, int tid) {
  const int lane = tid & 31, wid = tid >> 5;
  const int gcol = wid * 16 + (lane & 15);
  v8f acc = wmma_tile(Xs, kp, Wt, lane, gcol);
  const float bb = B[gcol];
  const int rb = (lane & 16) ? 8 : 0;
#pragma unroll
  for (int i = 0; i < 8; ++i) {
    float x = acc[i] + bb;
    x = x > 0.f ? x : 0.f;
    Xn[(rb + i) * 128 + gcol] = f2bf(x);
  }
}

__device__ __forceinline__ void layer_lin(const __bf16* Xs, int kp,
                                          const __bf16* Wt, const float* B,
                                          float* Yf, int tid) {
  const int lane = tid & 31, wid = tid >> 5;
  const int gcol = wid * 16 + (lane & 15);
  v8f acc = wmma_tile(Xs, kp, Wt, lane, gcol);
  const float bb = B[gcol];
  const int rb = (lane & 16) ? 8 : 0;
#pragma unroll
  for (int i = 0; i < 8; ++i)
    Yf[(rb + i) * 128 + gcol] = acc[i] + bb;
}

__device__ __forceinline__ void ln_stats(const float* Yf, float* St, int tid) {
  if (tid < 16) {
    float s = 0.f, ss = 0.f;
    const float* yr = Yf + tid * 128;
#pragma unroll 8
    for (int k = 0; k < 128; ++k) { float x = yr[k]; s += x; ss += x * x; }
    const float mu = s * (1.f / 128.f);
    const float var = ss * (1.f / 128.f) - mu * mu;
    St[tid] = mu;
    St[16 + tid] = rsqrtf(var + 1e-5f);
  }
}

// ---------------------------------------------------------------------------
// Weight conversion: W f32 [K][N] -> Wt bf16 [N][Kp] (K zero-padded to Kp).
// ---------------------------------------------------------------------------
__global__ __launch_bounds__(256) void k_convert_w(const float* __restrict__ W,
                                                   __bf16* __restrict__ Wt,
                                                   int K, int Kp, int N) {
  int t = blockIdx.x * 256 + threadIdx.x;
  if (t >= N * Kp) return;
  int n = t / Kp, k = t % Kp;
  float v = (k < K) ? W[(size_t)k * N + n] : 0.f;
  Wt[t] = f2bf(v);
}

// ---------------------------------------------------------------------------
// Mesh edge features -> padded bf16 rows + senders/receivers/mask.
// ---------------------------------------------------------------------------
__global__ __launch_bounds__(256) void k_mesh(const float* __restrict__ coords,
                                              const int* __restrict__ edges,
                                              const float* __restrict__ mean,
                                              const float* __restrict__ stdv,
                                              int* __restrict__ snd,
                                              int* __restrict__ rcv,
                                              float* __restrict__ msk,
                                              __bf16* __restrict__ xin, int E) {
  int e = blockIdx.x * 256 + threadIdx.x;
  if (e >= E) return;
  int s = edges[2 * e], r = edges[2 * e + 1];
  snd[e] = s; rcv[e] = r; msk[e] = 1.f;
  float r0 = coords[3 * s] - coords[3 * r];
  float r1 = coords[3 * s + 1] - coords[3 * r + 1];
  float r2 = coords[3 * s + 2] - coords[3 * r + 2];
  float d = sqrtf(r0 * r0 + r1 * r1 + r2 * r2);
  float f[4] = {r0, r1, r2, d};
  __bf16* xr = xin + (size_t)e * 32;
  for (int j = 0; j < 32; ++j) {
    float v = (j < 4) ? (f[j] - mean[j]) / stdv[j] : 0.f;
    xr[j] = f2bf(v);
  }
}

// ---------------------------------------------------------------------------
// kNN (K=8): one wave per query; per-lane register top-8, LDS merge by lane 0.
// Directly emits normalized*masked, 32-padded bf16 edge features.
// ---------------------------------------------------------------------------
__global__ __launch_bounds__(256) void k_knn(const float* __restrict__ coords,
                                             int Q, int T, int toff, float radius,
                                             const float* __restrict__ mean,
                                             const float* __restrict__ stdv,
                                             int fdim,
                                             const float* __restrict__ exA,
                                             const float* __restrict__ exB,
                                             int ebase,
                                             int* __restrict__ snd,
                                             int* __restrict__ rcv,
                                             float* __restrict__ msk,
                                             __bf16* __restrict__ xin) {
  __shared__ float sd[2048];
  __shared__ int   si[2048];
  const int tid = threadIdx.x, lane = tid & 31, wid = tid >> 5;
  const int q = blockIdx.x * 8 + wid;
  const bool valid = q < Q;
  float qx = 0.f, qy = 0.f, qz = 0.f;
  if (valid) { qx = coords[3 * q]; qy = coords[3 * q + 1]; qz = coords[3 * q + 2]; }
  float bd[8]; int bi[8];
#pragma unroll
  for (int j = 0; j < 8; ++j) { bd[j] = __builtin_inff(); bi[j] = 0; }
  if (valid) {
    for (int t = lane; t < T; t += 32) {
      const float* tp = coords + (size_t)(toff + t) * 3;
      float dx = qx - tp[0], dy = qy - tp[1], dz = qz - tp[2];
      float cd = dx * dx + dy * dy + dz * dz;
      int ci = t;
#pragma unroll
      for (int j = 0; j < 8; ++j) {        // sorted insert (bubble-down)
        if (cd < bd[j]) {
          float td = bd[j]; bd[j] = cd; cd = td;
          int ti = bi[j]; bi[j] = ci; ci = ti;
        }
      }
    }
  }
  const int base = wid * 256 + lane * 8;
#pragma unroll
  for (int j = 0; j < 8; ++j) { sd[base + j] = bd[j]; si[base + j] = bi[j]; }
  __syncthreads();
  if (valid && lane == 0) {
    const int wb = wid * 256;
    for (int p = 0; p < 8; ++p) {
      float best = __builtin_inff(); int bj = -1;
      for (int j = 0; j < 256; ++j) { float v = sd[wb + j]; if (v < best) { best = v; bj = j; } }
      int tsel = (bj >= 0) ? si[wb + bj] : 0;
      if (bj >= 0) sd[wb + bj] = __builtin_inff();
      int rnode = toff + tsel;
      float d = sqrtf(fmaxf(best, 0.f));
      float m = (d < radius) ? 1.f : 0.f;
      int eg = ebase + q * 8 + p;
      snd[eg] = q; rcv[eg] = rnode; msk[eg] = m;
      float f[6];
      f[0] = qx - coords[3 * rnode];
      f[1] = qy - coords[3 * rnode + 1];
      f[2] = qz - coords[3 * rnode + 2];
      f[3] = d;
      if (fdim == 6) { f[4] = *exA; f[5] = *exB; }
      __bf16* xr = xin + (size_t)eg * 32;
      for (int j = 0; j < 32; ++j) {
        float v = 0.f;
        if (j < fdim && m > 0.f) v = ((f[j] - mean[j]) / stdv[j]) * m;
        xr[j] = f2bf(v);
      }
    }
  }
}

// ---------------------------------------------------------------------------
// Node features: one-hot(9) + coords(3), normalized, padded to 32, bf16.
// ---------------------------------------------------------------------------
__global__ __launch_bounds__(256) void k_nodefeat(const float* __restrict__ coords,
                                                  const int* __restrict__ ntype,
                                                  const float* __restrict__ mean,
                                                  const float* __restrict__ stdv,
                                                  __bf16* __restrict__ xin, int N) {
  int n = blockIdx.x * 256 + threadIdx.x;
  if (n >= N) return;
  int ty = ntype[n];
  __bf16* xr = xin + (size_t)n * 32;
  for (int j = 0; j < 32; ++j) {
    float v = 0.f;
    if (j < 9)       v = (ty == j) ? 1.f : 0.f;
    else if (j < 12) v = coords[3 * n + (j - 9)];
    if (j < 12) v = (v - mean[j]) / stdv[j];
    xr[j] = f2bf(v);
  }
}

// ---------------------------------------------------------------------------
// Generic 3-layer MLP + LN encoder: Xin bf16 [M][k0p] -> out f32 + bf16.
// Block = 256 threads (8 waves) = one 16-row x 128-col tile.
// ---------------------------------------------------------------------------
__global__ __launch_bounds__(256) void k_mlp_enc(
    const __bf16* __restrict__ Xin, int M, int k0p,
    const __bf16* __restrict__ W1, const float* __restrict__ B1,
    const __bf16* __restrict__ W2, const float* __restrict__ B2,
    const __bf16* __restrict__ W3, const float* __restrict__ B3,
    const float* __restrict__ G, const float* __restrict__ Bt,
    float* __restrict__ outF, __bf16* __restrict__ outB) {
  extern __shared__ __align__(16) char smem[];
  __bf16* Xs = (__bf16*)smem;
  __bf16* X1 = (__bf16*)(smem + (size_t)32 * k0p);
  __bf16* X2 = X1 + 16 * 128;
  float*  Yf = (float*)(X2 + 16 * 128);
  float*  St = Yf + 16 * 128;
  const int tid = threadIdx.x;
  const int mb = blockIdx.x * 16;
  const int totin = 16 * k0p;
  for (int idx = tid * 8; idx < totin; idx += 2048) {
    int r = idx / k0p, c = idx % k0p;
    int gr = mb + r;
    ull2 v = {0ull, 0ull};
    if (gr < M) v = *(const ull2*)(Xin + (size_t)gr * k0p + c);
    *(ull2*)(Xs + idx) = v;
  }
  __syncthreads();
  layer_relu(Xs, k0p, W1, B1, X1, tid);  __syncthreads();
  layer_relu(X1, 128, W2, B2, X2, tid);  __syncthreads();
  layer_lin (X2, 128, W3, B3, Yf, tid);  __syncthreads();
  ln_stats(Yf, St, tid);                 __syncthreads();
  for (int idx = tid; idx < 2048; idx += 256) {
    int r = idx >> 7, c = idx & 127;
    int gr = mb + r;
    if (gr < M) {
      float y = (Yf[idx] - St[r]) * St[16 + r] * G[c] + Bt[c];
      outF[(size_t)gr * 128 + c] = y;
      outB[(size_t)gr * 128 + c] = f2bf(y);
    }
  }
}

// ---------------------------------------------------------------------------
// Processor edge MLP: gather [ef | v[s] | v[r]] (K=384) -> MLP+LN ->
// masked residual into ef (f32+bf16) + f32 atomic scatter into agg[recv].
// ---------------------------------------------------------------------------
__global__ __launch_bounds__(256) void k_proc_edge(
    int E, const int* __restrict__ snd, const int* __restrict__ rcv,
    const float* __restrict__ msk,
    float* __restrict__ efF, __bf16* __restrict__ efB,
    const __bf16* __restrict__ vB,
    const __bf16* __restrict__ W1, const float* __restrict__ B1,
    const __bf16* __restrict__ W2, const float* __restrict__ B2,
    const __bf16* __restrict__ W3, const float* __restrict__ B3,
    const float* __restrict__ G, const float* __restrict__ Bt,
    float* __restrict__ agg, int aggoff) {
  extern __shared__ __align__(16) char smem[];
  __bf16* Xs = (__bf16*)smem;                       // 16 x 384
  __bf16* X1 = (__bf16*)(smem + (size_t)32 * 384);
  __bf16* X2 = X1 + 16 * 128;
  float*  Yf = (float*)(X2 + 16 * 128);
  float*  St = Yf + 16 * 128;
  const int tid = threadIdx.x;
  const int eb = blockIdx.x * 16;
  for (int idx = tid * 8; idx < 16 * 384; idx += 2048) {
    int r = idx / 384, c = idx % 384;
    int e = eb + r;
    ull2 v = {0ull, 0ull};
    if (e < E) {
      const __bf16* src;
      if (c < 128) src = efB + (size_t)e * 128 + c;
      else {
        int nd = (c < 256) ? snd[e] : rcv[e];
        src = vB + (size_t)nd * 128 + (c & 127);
      }
      v = *(const ull2*)src;
    }
    *(ull2*)(Xs + idx) = v;
  }
  __syncthreads();
  layer_relu(Xs, 384, W1, B1, X1, tid);  __syncthreads();
  layer_relu(X1, 128, W2, B2, X2, tid);  __syncthreads();
  layer_lin (X2, 128, W3, B3, Yf, tid);  __syncthreads();
  ln_stats(Yf, St, tid);                 __syncthreads();
  for (int idx = tid; idx < 2048; idx += 256) {
    int r = idx >> 7, c = idx & 127;
    int e = eb + r;
    if (e < E) {
      float y = (Yf[idx] - St[r]) * St[16 + r] * G[c] + Bt[c];
      float m = msk[e];
      float nv = efF[(size_t)e * 128 + c] + y * m;
      efF[(size_t)e * 128 + c] = nv;
      efB[(size_t)e * 128 + c] = f2bf(nv);
      atomicAdd(agg + (size_t)rcv[e] * 512 + aggoff + c, nv * m);
    }
  }
}

// ---------------------------------------------------------------------------
// Processor node MLP: [v | agg(4x128 f32->bf16)] (K=640) -> MLP+LN -> v += y.
// ---------------------------------------------------------------------------
__global__ __launch_bounds__(256) void k_proc_node(
    int M,
    float* __restrict__ vF, __bf16* __restrict__ vB,
    const float* __restrict__ agg,
    const __bf16* __restrict__ W1, const float* __restrict__ B1,
    const __bf16* __restrict__ W2, const float* __restrict__ B2,
    const __bf16* __restrict__ W3, const float* __restrict__ B3,
    const float* __restrict__ G, const float* __restrict__ Bt) {
  extern __shared__ __align__(16) char smem[];
  __bf16* Xs = (__bf16*)smem;                       // 16 x 640
  __bf16* X1 = (__bf16*)(smem + (size_t)32 * 640);
  __bf16* X2 = X1 + 16 * 128;
  float*  Yf = (float*)(X2 + 16 * 128);
  float*  St = Yf + 16 * 128;
  const int tid = threadIdx.x;
  const int nb = blockIdx.x * 16;
  for (int idx = tid * 8; idx < 16 * 640; idx += 2048) {
    int r = idx / 640, c = idx % 640;
    int n = nb + r;
    ull2 v = {0ull, 0ull};
    if (n < M) {
      if (c < 128) {
        v = *(const ull2*)(vB + (size_t)n * 128 + c);
      } else {
        const float* a = agg + (size_t)n * 512 + (c - 128);
        BF16x8 t;
#pragma unroll
        for (int j = 0; j < 8; ++j) t.h[j] = f2bf(a[j]);
        v = t.q;
      }
    }
    *(ull2*)(Xs + idx) = v;
  }
  __syncthreads();
  layer_relu(Xs, 640, W1, B1, X1, tid);  __syncthreads();
  layer_relu(X1, 128, W2, B2, X2, tid);  __syncthreads();
  layer_lin (X2, 128, W3, B3, Yf, tid);  __syncthreads();
  ln_stats(Yf, St, tid);                 __syncthreads();
  for (int idx = tid; idx < 2048; idx += 256) {
    int r = idx >> 7, c = idx & 127;
    int n = nb + r;
    if (n < M) {
      float y = (Yf[idx] - St[r]) * St[16 + r] * G[c] + Bt[c];
      float nv = vF[(size_t)n * 128 + c] + y;
      vF[(size_t)n * 128 + c] = nv;
      vB[(size_t)n * 128 + c] = f2bf(nv);
    }
  }
}

// ---------------------------------------------------------------------------
// Decoder: 128 -relu-> 128 -relu-> (WMMA) then scalar 128->3 final layer.
// ---------------------------------------------------------------------------
__global__ __launch_bounds__(256) void k_mlp_dec(
    const __bf16* __restrict__ vB, int M,
    const __bf16* __restrict__ W1, const float* __restrict__ B1,
    const __bf16* __restrict__ W2, const float* __restrict__ B2,
    const float* __restrict__ W3f, const float* __restrict__ B3f,
    float* __restrict__ out) {
  extern __shared__ __align__(16) char smem[];
  __bf16* Xs = (__bf16*)smem;        // 16 x 128
  __bf16* X1 = Xs + 16 * 128;
  __bf16* X2 = X1 + 16 * 128;
  const int tid = threadIdx.x;
  const int mb = blockIdx.x * 16;
  for (int idx = tid * 8; idx < 2048; idx += 2048) {
    int r = idx >> 7, c = idx & 127;
    int gr = mb + r;
    ull2 v = {0ull, 0ull};
    if (gr < M) v = *(const ull2*)(vB + (size_t)gr * 128 + c);
    *(ull2*)(Xs + idx) = v;
  }
  __syncthreads();
  layer_relu(Xs, 128, W1, B1, X1, tid);  __syncthreads();
  layer_relu(X1, 128, W2, B2, X2, tid);  __syncthreads();
  if (tid < 48) {
    int r = tid / 3, c = tid % 3;
    int gr = mb + r;
    if (gr < M) {
      float s = B3f[c];
      for (int k = 0; k < 128; ++k) s += bf2f(X2[r * 128 + k]) * W3f[k * 3 + c];
      out[(size_t)gr * 3 + c] = s;
    }
  }
}

// ---------------------------------------------------------------------------
// Host side.
// Input leaf order assumption: JAX pytree flatten (dict keys sorted at every
// level, lists/tuples in order):
//   0 F_press, 1 coords, 2 depth, 3 edges, 4 friction, 5 node_type,
//   6..15 norm_stats {die,edge,holder,node,punch} x (mean,std),
//   16..21 dec layers (W,b)x3,
//   22..53 enc_edge[0..3] x {W,b x3, ln_g, ln_b},
//   54..61 enc_node,
//   62..   proc[s] x { edges[0..3] x 8 leaves, node x 8 leaves } (40/step).
// ---------------------------------------------------------------------------
namespace {
enum {
  L_FPRESS = 0, L_COORDS = 1, L_DEPTH = 2, L_EDGES = 3, L_FRICTION = 4,
  L_NODETYPE = 5,
  L_NS_DIE_M = 6, L_NS_DIE_S = 7, L_NS_EDGE_M = 8, L_NS_EDGE_S = 9,
  L_NS_HOLD_M = 10, L_NS_HOLD_S = 11, L_NS_NODE_M = 12, L_NS_NODE_S = 13,
  L_NS_PUNCH_M = 14, L_NS_PUNCH_S = 15,
  L_DEC = 16,       // 6 leaves
  L_ENC_EDGE = 22,  // 4 * 8 leaves
  L_ENC_NODE = 54,  // 8 leaves
  L_PROC = 62       // 15 * 40 leaves
};
struct MlpW {
  const __bf16* w1; const __bf16* w2; const __bf16* w3;
  const float* b1; const float* b2; const float* b3;
  const float* g; const float* bt;
};
}  // namespace

extern "C" void kernel_launch(void* const* d_in, const int* in_sizes, int n_in,
                              void* d_out, int out_size, void* d_ws, size_t ws_size,
                              hipStream_t stream) {
  (void)in_sizes; (void)n_in; (void)out_size; (void)ws_size;

  const int N = 20000;
  const int EM = 30000;
  const int Q = 4981;           // OFF_PUNCH
  const int EK = Q * 8;         // 39848 edges per kNN type
  const int ET = EM + 3 * EK;   // 149544 total edges
  const int STEPS = 15;

  // ---- workspace bump allocator -------------------------------------------
  char* wsb = (char*)d_ws;
  size_t cur = 0;
  auto alloc = [&](size_t bytes) -> void* {
    void* p = wsb + cur;
    cur = (cur + bytes + 255) & ~(size_t)255;
    return p;
  };
  float*  vF   = (float*)  alloc((size_t)N * 128 * 4);
  __bf16* vB   = (__bf16*) alloc((size_t)N * 128 * 2);
  float*  agg  = (float*)  alloc((size_t)N * 512 * 4);
  float*  efF  = (float*)  alloc((size_t)ET * 128 * 4);
  __bf16* efB  = (__bf16*) alloc((size_t)ET * 128 * 2);
  __bf16* xinE = (__bf16*) alloc((size_t)ET * 32 * 2);
  __bf16* xinN = (__bf16*) alloc((size_t)N * 32 * 2);
  int*    snd  = (int*)    alloc((size_t)ET * 4);
  int*    rcv  = (int*)    alloc((size_t)ET * 4);
  float*  msk  = (float*)  alloc((size_t)ET * 4);

  // ---- weight conversion (f32 [K][N] -> bf16 [N][Kp]) ---------------------
  auto regW = [&](int leaf, int K, int Nout) -> const __bf16* {
    int Kp = (K + 31) & ~31;
    __bf16* dst = (__bf16*)alloc((size_t)Nout * Kp * 2);
    int tot = Nout * Kp;
    k_convert_w<<<(tot + 255) / 256, 256, 0, stream>>>(
        (const float*)d_in[leaf], dst, K, Kp, Nout);
    return dst;
  };
  auto regMlp = [&](int base, int k0) -> MlpW {
    MlpW m;
    m.w1 = regW(base + 0, k0, 128);  m.b1 = (const float*)d_in[base + 1];
    m.w2 = regW(base + 2, 128, 128); m.b2 = (const float*)d_in[base + 3];
    m.w3 = regW(base + 4, 128, 128); m.b3 = (const float*)d_in[base + 5];
    m.g  = (const float*)d_in[base + 6];
    m.bt = (const float*)d_in[base + 7];
    return m;
  };

  const int edims[4] = {4, 6, 4, 6};
  MlpW encNode = regMlp(L_ENC_NODE, 12);
  MlpW encEdge[4];
  for (int i = 0; i < 4; ++i) encEdge[i] = regMlp(L_ENC_EDGE + 8 * i, edims[i]);
  const __bf16* decW1 = regW(L_DEC + 0, 128, 128);
  const float*  decB1 = (const float*)d_in[L_DEC + 1];
  const __bf16* decW2 = regW(L_DEC + 2, 128, 128);
  const float*  decB2 = (const float*)d_in[L_DEC + 3];
  const float*  decW3 = (const float*)d_in[L_DEC + 4];   // f32 (128,3)
  const float*  decB3 = (const float*)d_in[L_DEC + 5];
  MlpW procE[STEPS][4];
  MlpW procN[STEPS];
  for (int s = 0; s < STEPS; ++s) {
    int base = L_PROC + 40 * s;
    for (int i = 0; i < 4; ++i) procE[s][i] = regMlp(base + 8 * i, 384);
    procN[s] = regMlp(base + 32, 640);
  }

  const float* coords = (const float*)d_in[L_COORDS];

  // ---- graph construction: features ---------------------------------------
  k_mesh<<<(EM + 255) / 256, 256, 0, stream>>>(
      coords, (const int*)d_in[L_EDGES],
      (const float*)d_in[L_NS_EDGE_M], (const float*)d_in[L_NS_EDGE_S],
      snd, rcv, msk, xinE, EM);

  // kNN edge types: punch(2.5), die(1.5), holder(1.5)
  struct KnnCfg { int toff, T, fdim, stM, stS, ebase; float radius;
                  const float *exA, *exB; };
  const KnnCfg kc[3] = {
    {4981, 6060, 6, L_NS_PUNCH_M, L_NS_PUNCH_S, EM,          2.5f,
     (const float*)d_in[L_FRICTION], (const float*)d_in[L_DEPTH]},
    {11041, 5694, 4, L_NS_DIE_M, L_NS_DIE_S,    EM + EK,     1.5f,
     nullptr, nullptr},
    {16735, 3265, 6, L_NS_HOLD_M, L_NS_HOLD_S,  EM + 2 * EK, 1.5f,
     (const float*)d_in[L_FRICTION], (const float*)d_in[L_FPRESS]},
  };
  for (int i = 0; i < 3; ++i) {
    k_knn<<<(Q + 7) / 8, 256, 0, stream>>>(
        coords, Q, kc[i].T, kc[i].toff, kc[i].radius,
        (const float*)d_in[kc[i].stM], (const float*)d_in[kc[i].stS],
        kc[i].fdim, kc[i].exA, kc[i].exB, kc[i].ebase, snd, rcv, msk, xinE);
  }

  k_nodefeat<<<(N + 255) / 256, 256, 0, stream>>>(
      coords, (const int*)d_in[L_NODETYPE],
      (const float*)d_in[L_NS_NODE_M], (const float*)d_in[L_NS_NODE_S],
      xinN, N);

  auto smemFor = [](int k0p) -> size_t {
    return (size_t)32 * k0p + 4096 + 4096 + 8192 + 128;
  };

  // ---- encoders (WMMA) ----------------------------------------------------
  k_mlp_enc<<<(N + 15) / 16, 256, smemFor(32), stream>>>(
      xinN, N, 32, encNode.w1, encNode.b1, encNode.w2, encNode.b2,
      encNode.w3, encNode.b3, encNode.g, encNode.bt, vF, vB);

  const int eCnt[4]  = {EM, EK, EK, EK};
  const int eBase[4] = {0, EM, EM + EK, EM + 2 * EK};
  for (int i = 0; i < 4; ++i) {
    int E = eCnt[i], b = eBase[i];
    k_mlp_enc<<<(E + 15) / 16, 256, smemFor(32), stream>>>(
        xinE + (size_t)b * 32, E, 32,
        encEdge[i].w1, encEdge[i].b1, encEdge[i].w2, encEdge[i].b2,
        encEdge[i].w3, encEdge[i].b3, encEdge[i].g, encEdge[i].bt,
        efF + (size_t)b * 128, efB + (size_t)b * 128);
  }

  // ---- 15 message-passing steps (WMMA everywhere) -------------------------
  for (int s = 0; s < STEPS; ++s) {
    hipMemsetAsync(agg, 0, (size_t)N * 512 * 4, stream);
    for (int i = 0; i < 4; ++i) {
      int E = eCnt[i], b = eBase[i];
      const MlpW& m = procE[s][i];
      k_proc_edge<<<(E + 15) / 16, 256, smemFor(384), stream>>>(
          E, snd + b, rcv + b, msk + b,
          efF + (size_t)b * 128, efB + (size_t)b * 128, vB,
          m.w1, m.b1, m.w2, m.b2, m.w3, m.b3, m.g, m.bt,
          agg, 128 * i);
    }
    const MlpW& m = procN[s];
    k_proc_node<<<(N + 15) / 16, 256, smemFor(640), stream>>>(
        N, vF, vB, agg,
        m.w1, m.b1, m.w2, m.b2, m.w3, m.b3, m.g, m.bt);
  }

  // ---- decoder ------------------------------------------------------------
  k_mlp_dec<<<(N + 15) / 16, 256, (size_t)16 * 128 * 2 * 3, stream>>>(
      vB, N, decW1, decB1, decW2, decB2, decW3, decB3, (float*)d_out);
}
